// LineAttention_10874857193780
// MI455X (gfx1250) — matrix operand order repaired and checked by reading, not compiled, not run on hardware
//
#include <hip/hip_runtime.h>
#include <hip/hip_bf16.h>

// ---------------------------------------------------------------------------
// Line attention, CDNA5 (gfx1250), bf16 WMMA (v_wmma_f32_16x16x32_bf16),
// TDM (tensor_load_to_lds) double-buffered weight staging.
//   N=196608 tokens, DIM=256, LINE=24, HEADS=8, DHEAD=64, INNER=512.
// One workgroup (256 thr = 8 waves) handles LB=4 lines (M=96 rows = 6 tiles).
// Weights converted once to bf16 [N][K] (B-fragment layout) in workspace.
// ---------------------------------------------------------------------------

#define DIM   256
#define LINE  24
#define HEADS 8
#define DHEAD 64
#define INNER 512
#define LB    4
#define MROWS (LB * LINE)   // 96
#define NTOK  196608

// LDS strides in bf16 elements: multiples of 8 (16B fragment alignment),
// not multiples of 128 (stagger rows across the 64 LDS banks).
#define XS  264   // x tile        [96][264]
#define WS_ 264   // staged weight [2][64][264]
#define QS  72    // q / k         [104][72]
#define VTS 104   // v^T           [64][104]
#define AS  40    // attn probs    [32][40]  (bf16)
#define SS  33    // sim           [32][33]  (f32)
#define OS  520   // head-concat   [96][520]

typedef __attribute__((ext_vector_type(16))) __bf16    v16bf;
typedef __attribute__((ext_vector_type(8)))  float     v8f;
typedef __attribute__((ext_vector_type(4)))  unsigned  v4u;
typedef __attribute__((ext_vector_type(4)))  int       v4i;
typedef __attribute__((ext_vector_type(8)))  int       v8i;

union FragU { v16bf v; uint4 q[2]; };

static __device__ __forceinline__ unsigned short f2bf(float f) {
  unsigned u = __builtin_bit_cast(unsigned, f);
  u += 0x7FFFu + ((u >> 16) & 1u);               // round-to-nearest-even
  return (unsigned short)(u >> 16);
}

static __device__ __forceinline__ v8f wmma_bf16(v16bf a, v16bf b, v8f c) {
  // (neg_a, A, neg_b, B, c_mod, C, reuse_a, reuse_b)
  return __builtin_amdgcn_wmma_f32_16x16x32_bf16(false, a, false, b,
                                                 (short)0, c, false, false);
}

// A fragment (16x32 bf16, row-major source):
//   lanes 0-15 : K 0-7 in elems 0-7, K 16-23 in elems 8-15
//   lanes 16-31: K 8-15,            K 24-31
static __device__ __forceinline__ v16bf loadA(const unsigned short* base, int row,
                                              int kb, int stride, int lhalf, int half) {
  const unsigned short* p = base + (row + lhalf) * stride + kb + half * 8;
  FragU f;
  f.q[0] = *(const uint4*)p;
  f.q[1] = *(const uint4*)(p + 16);
  return f.v;
}

// B fragment (32x16 bf16) from W^T stored [N][K]:
//   lane = column N, lanes 0-15 hold K 0-7,16-23, lanes 16-31 hold K 8-15,24-31.
static __device__ __forceinline__ v16bf loadB(const unsigned short* base, int col,
                                              int kb, int stride, int lhalf, int half) {
  const unsigned short* p = base + (col + lhalf) * stride + kb + half * 8;
  FragU f;
  f.q[0] = *(const uint4*)p;
  f.q[1] = *(const uint4*)(p + 16);
  return f.v;
}

// ---------------------------------------------------------------------------
// TDM: DMA one 64x256 bf16 weight slice (global, row-major, stride 256) into
// LDS, inserting 4 DWORDs of padding after every 128 DWORDs (one 512B row)
// so the LDS image has row stride WS_=264 bf16. 2D descriptor, tracked by
// TENSORcnt. Issued by wave 0 only (EXEC is ignored by TDM).
// ---------------------------------------------------------------------------
static __device__ __forceinline__ void tdm_stage_weight(const unsigned short* gsrc,
                                                        unsigned lds_byte) {
  unsigned long long ga = (unsigned long long)(size_t)gsrc;
  v4u g0 = (v4u){ 1u,                                   // count=1 (valid D#)
                  lds_byte,                             // lds_addr
                  (unsigned)ga,                         // global_addr[31:0]
                  ((unsigned)(ga >> 32) & 0x01FFFFFFu)  // global_addr[56:32]
                      | (2u << 30) };                   // type=2 ("image")
  v8i g1 = (v8i){ (int)((1u << 16)      // data_size = 2 bytes
                      | (1u << 20)      // pad_enable
                      | (6u << 22)      // pad_interval: 2^(6+1)=128 DWORDs
                      | (3u << 25)),    // pad_amount : 3+1 = 4 DWORDs
                  (int)(256u << 16),    // tensor_dim0 = 256   (bits 79:48)
                  (int)(64u  << 16),    // tensor_dim1 = 64    (bits 111:80)
                  (int)(256u << 16),    // tile_dim0   = 256   (bits 127:112)
                  64,                   // tile_dim1=64, tile_dim2=0
                  256,                  // tensor_dim0_stride = 256
                  0, 0 };
  v4i gz = (v4i){0, 0, 0, 0};
#if __has_include(<hip/amd_detail/amd_gfx1250_TDM.h>)
  v8i gz8 = (v8i){0, 0, 0, 0, 0, 0, 0, 0};
  __builtin_amdgcn_tensor_load_to_lds(g0, g1, gz, gz, gz8, 0);
#else
  __builtin_amdgcn_tensor_load_to_lds(g0, g1, gz, gz, 0);
#endif
}

// ---------------------------------------------------------------------------
// Prep: fp32 weights -> bf16, transposed to [N][K] row-major.
//   wqT/wkT/wvT: [512][256], woT: [256][512]
// ---------------------------------------------------------------------------
__global__ void la_prep(const float* __restrict__ Wq, const float* __restrict__ Wkv,
                        const float* __restrict__ Wout,
                        unsigned short* __restrict__ wqT, unsigned short* __restrict__ wkT,
                        unsigned short* __restrict__ wvT, unsigned short* __restrict__ woT) {
  int idx = blockIdx.x * 256 + threadIdx.x;
  if (idx >= INNER * DIM) return;
  int n = idx >> 8, k = idx & 255;               // [512][256]
  wqT[idx] = f2bf(Wq [k * INNER + n]);
  wkT[idx] = f2bf(Wkv[k * 2 * INNER + n]);
  wvT[idx] = f2bf(Wkv[k * 2 * INNER + INNER + n]);
  int n2 = idx >> 9, k2 = idx & 511;             // [256][512]
  woT[idx] = f2bf(Wout[k2 * DIM + n2]);
}

// ---------------------------------------------------------------------------
// Main fused kernel: 2048 blocks x 256 threads, 4 lines per block.
// ---------------------------------------------------------------------------
__global__ __launch_bounds__(256) void la_main(
    const float* __restrict__ x,
    const unsigned short* __restrict__ wqT, const unsigned short* __restrict__ wkT,
    const unsigned short* __restrict__ wvT, const unsigned short* __restrict__ woT,
    const float* __restrict__ b_out, const float* __restrict__ pos_emb,
    float* __restrict__ out) {

  __shared__ __align__(16) unsigned short s_x [MROWS * XS];
  __shared__ __align__(16) unsigned short s_w [2][64 * WS_];
  __shared__ __align__(16) unsigned short s_q [(MROWS + 8) * QS];
  __shared__ __align__(16) unsigned short s_k [(MROWS + 8) * QS];
  __shared__ __align__(16) unsigned short s_vT[DHEAD * VTS];
  __shared__ __align__(16) unsigned short s_at[LB][32 * AS];
  __shared__ __align__(16) unsigned short s_oc[MROWS * OS];
  __shared__ float s_sim[LB][32 * SS];

  const int tid   = threadIdx.x;
  const int widx  = tid >> 5;
  const int lane  = tid & 31;
  const int half  = lane >> 4;
  const int lhalf = lane & 15;

  const unsigned short* wmats[3] = { wqT, wkT, wvT };

  // Kick off the DMA of the first weight slice before anything else.
  if (widx == 0)
    tdm_stage_weight(wmats[0], (unsigned)(size_t)&s_w[0][0]);

  // ---- P0: load x tile (fp32 -> bf16) + zero pad rows -------------------
  {
    const float4* xg = (const float4*)(x + (size_t)blockIdx.x * MROWS * DIM);
    for (int idx = tid; idx < MROWS * (DIM / 4); idx += 256) {
      int row = idx >> 6, c4 = idx & 63;
      float4 v = xg[idx];
      unsigned* d = (unsigned*)&s_x[row * XS + c4 * 4];
      d[0] = (unsigned)f2bf(v.x) | ((unsigned)f2bf(v.y) << 16);
      d[1] = (unsigned)f2bf(v.z) | ((unsigned)f2bf(v.w) << 16);
    }
    for (int idx = tid; idx < 8 * QS; idx += 256) {       // q/k pad rows 96..103
      s_q[MROWS * QS + idx] = 0;
      s_k[MROWS * QS + idx] = 0;
    }
    for (int idx = tid; idx < DHEAD * 8; idx += 256) {    // v^T pad cols 96..103
      s_vT[(idx >> 3) * VTS + MROWS + (idx & 7)] = 0;
    }
    // Warm L2/WGP$ with this wave's Wout slice for P3 (global_prefetch_b8:
    // each enabled lane prefetches its own cacheline).
    for (int it = 0; it < 2; ++it) {
      int nt = widx * 2 + it;
      for (int c = 0; c < 4; ++c)
        __builtin_prefetch(&woT[(nt * 16 + lhalf) * INNER + half * 256 + c * 64], 0, 1);
    }
  }
  __syncthreads();

  const int w   = widx & 3;        // line handled in attention phases
  const int smt = widx >> 2;       // M-tile within the line
  const int r0  = w * LINE;

  // ---- per-head loop ----------------------------------------------------
  for (int h = 0; h < HEADS; ++h) {
    // P1: q / k / v projections; weight slice DMA'd by TDM, double buffered.
    for (int mat = 0; mat < 3; ++mat) {
      const int s = h * 3 + mat;
      if (widx == 0)
        __builtin_amdgcn_s_wait_tensorcnt(0);   // slice for step s is in LDS
      __syncthreads();                           // publish; prior reads retired
      if (widx == 0 && s + 1 < 3 * HEADS) {      // overlap next DMA w/ compute
        int m1 = (s + 1) % 3, h1 = (s + 1) / 3;
        tdm_stage_weight(wmats[m1] + (size_t)h1 * DHEAD * DIM,
                         (unsigned)(size_t)&s_w[(s + 1) & 1][0]);
      }
      const unsigned short* wbuf = &s_w[s & 1][0];

      // 6 Mtiles x 4 Ntiles = 24 tiles, 3 per wave
      for (int jj = 0; jj < 3; ++jj) {
        int tt = widx * 3 + jj;
        int mt = tt >> 2, nt = tt & 3;
        v8f acc = {};
#pragma unroll
        for (int kk = 0; kk < 8; ++kk)
          acc = wmma_bf16(loadA(s_x, mt * 16, kk * 32, XS, lhalf, half),
                          loadB(wbuf, nt * 16, kk * 32, WS_, lhalf, half), acc);
        int mrow = mt * 16 + half * 8;
        int ncol = nt * 16 + lhalf;
        if (mat == 0) {
          unsigned short* d = &s_q[mrow * QS + ncol];
#pragma unroll
          for (int rr = 0; rr < 8; ++rr) d[rr * QS] = f2bf(acc[rr] * 0.125f); // 1/sqrt(64)
        } else if (mat == 1) {
          unsigned short* d = &s_k[mrow * QS + ncol];
#pragma unroll
          for (int rr = 0; rr < 8; ++rr) d[rr * QS] = f2bf(acc[rr]);
        } else {                                            // v stored transposed
          unsigned* d = (unsigned*)&s_vT[ncol * VTS + mrow];
#pragma unroll
          for (int rr = 0; rr < 4; ++rr)
            d[rr] = (unsigned)f2bf(acc[2 * rr]) | ((unsigned)f2bf(acc[2 * rr + 1]) << 16);
        }
      }
    }
    __syncthreads();                             // q/k/vT complete

    // P2a: sim = q @ k^T  (wave -> (line, Mtile), both N tiles)
    for (int nt = 0; nt < 2; ++nt) {
      v8f acc = {};
#pragma unroll
      for (int kk = 0; kk < 2; ++kk)
        acc = wmma_bf16(loadA(s_q, r0 + smt * 16, kk * 32, QS, lhalf, half),
                        loadB(s_k, r0 + nt * 16, kk * 32, QS, lhalf, half), acc);
      float* ds = &s_sim[w][(smt * 16 + half * 8) * SS + nt * 16 + lhalf];
#pragma unroll
      for (int rr = 0; rr < 8; ++rr) ds[rr * SS] = acc[rr];
    }
    __syncthreads();

    // P2b: softmax (+pos bias), one lane per row; pad rows/cols -> exact 0.
    if (tid < LB * 32) {
      int sw = tid >> 5, i = tid & 31;
      unsigned* arow = (unsigned*)&s_at[sw][i * AS];
      if (i < LINE) {
        const float* pe = pos_emb + (h * LINE + i) * LINE;
        float r[LINE], mx = -1e30f;
        for (int j = 0; j < LINE; ++j) {
          r[j] = s_sim[sw][i * SS + j] + pe[j];
          mx = fmaxf(mx, r[j]);
        }
        float s = 0.f;
        for (int j = 0; j < LINE; ++j) { r[j] = __expf(r[j] - mx); s += r[j]; }
        float inv = 1.0f / s;
        for (int j = 0; j < LINE / 2; ++j)
          arow[j] = (unsigned)f2bf(r[2 * j] * inv) | ((unsigned)f2bf(r[2 * j + 1] * inv) << 16);
        for (int j = LINE / 2; j < 16; ++j) arow[j] = 0;   // cols 24..31 := 0
      } else {
        for (int j = 0; j < 16; ++j) arow[j] = 0;          // rows 24..31 := 0
      }
    }
    __syncthreads();

    // P2c: out_h = attn @ v   (single K=32 step; pad cols of attn are zero)
    {
      v16bf aat = loadA(&s_at[w][0], smt * 16, 0, AS, lhalf, half);
      int lm = smt * 16 + half * 8;                        // local row in line
      for (int nt = 0; nt < 4; ++nt) {
        v8f acc = {};
        acc = wmma_bf16(aat, loadB(s_vT, nt * 16, r0, VTS, lhalf, half), acc);
        unsigned short* doc = &s_oc[(r0 + lm) * OS + h * DHEAD + nt * 16 + lhalf];
#pragma unroll
        for (int rr = 0; rr < 8; ++rr)
          if (lm + rr < LINE) doc[rr * OS] = f2bf(acc[rr]);
      }
    }
    __syncthreads();                             // s_oc cols of head h done
  }

  // ---- P3: out = concat(heads) @ Wout + b_out ---------------------------
  for (int it = 0; it < 2; ++it) {
    int nt = widx * 2 + it;                                // 16 N-tiles / 8 waves
    float bias = b_out[nt * 16 + lhalf];
    v8f acc[6];
#pragma unroll
    for (int mt = 0; mt < 6; ++mt)
#pragma unroll
      for (int rr = 0; rr < 8; ++rr) acc[mt][rr] = bias;
    for (int kk = 0; kk < 16; ++kk) {
      v16bf bfr = loadB(woT, nt * 16, kk * 32, INNER, lhalf, half);
#pragma unroll
      for (int mt = 0; mt < 6; ++mt)
        acc[mt] = wmma_bf16(loadA(s_oc, mt * 16, kk * 32, OS, lhalf, half), bfr, acc[mt]);
    }
    float* og = out + (size_t)blockIdx.x * MROWS * DIM + nt * 16 + lhalf;
#pragma unroll
    for (int mt = 0; mt < 6; ++mt)
#pragma unroll
      for (int rr = 0; rr < 8; ++rr)
        og[(size_t)(mt * 16 + half * 8 + rr) * DIM] = acc[mt][rr];
  }
}

// ---------------------------------------------------------------------------
extern "C" void kernel_launch(void* const* d_in, const int* in_sizes, int n_in,
                              void* d_out, int out_size, void* d_ws, size_t ws_size,
                              hipStream_t stream) {
  (void)in_sizes; (void)n_in; (void)out_size; (void)ws_size;
  const float* x     = (const float*)d_in[0];
  const float* Wq    = (const float*)d_in[1];
  const float* Wkv   = (const float*)d_in[2];
  const float* Wout  = (const float*)d_in[3];
  const float* b_out = (const float*)d_in[4];
  const float* pose  = (const float*)d_in[5];

  unsigned short* wqT = (unsigned short*)d_ws;             // [512][256] bf16
  unsigned short* wkT = wqT + INNER * DIM;
  unsigned short* wvT = wkT + INNER * DIM;
  unsigned short* woT = wvT + INNER * DIM;                 // [256][512] bf16

  la_prep<<<(INNER * DIM + 255) / 256, 256, 0, stream>>>(Wq, Wkv, Wout, wqT, wkT, wvT, woT);
  la_main<<<NTOK / MROWS, 256, 0, stream>>>(x, wqT, wkT, wvT, woT, b_out, pose, (float*)d_out);
}
